// Head_50732153701071
// MI455X (gfx1250) — matrix operand order repaired and checked by reading, not compiled, not run on hardware
//
#include <hip/hip_runtime.h>

// Problem constants (reference: B,T,C,H = 16,2048,1024,64)
constexpr int Bn = 16;
constexpr int Tn = 2048;
constexpr int Cn = 1024;
constexpr int Hn = 64;

typedef __attribute__((ext_vector_type(16))) _Float16 v16h;
typedef __attribute__((ext_vector_type(8)))  _Float16 v8h;
typedef __attribute__((ext_vector_type(8)))  float    v8f;
typedef __attribute__((ext_vector_type(4)))  unsigned int v4u;
typedef __attribute__((ext_vector_type(4)))  int      v4i;
typedef __attribute__((ext_vector_type(8)))  int      v8i;

// ---------------------------------------------------------------------------
// TDM descriptor build + issue (CDNA5 ISA).
// 2D tile of 16-bit elements: tile_dim0 contiguous, tile_dim1 rows with
// tensor_dim0_stride between rows. LDS padding: pad_amount DWORDs inserted
// after every pad_interval DWORDs (codes per D# group1).
// ---------------------------------------------------------------------------
__device__ __forceinline__ void tdm_load_tile(unsigned lds_off, const void* gptr,
                                              unsigned flags_w0,
                                              unsigned tile_d0, unsigned tile_d1,
                                              unsigned long long stride_elems) {
  const unsigned long long ga = (unsigned long long)(uintptr_t)gptr;
  v4u g0 = { 1u,                                   // count=1 (valid, user mode)
             lds_off,                              // lds_addr (bytes)
             (unsigned)(ga & 0xFFFFFFFFull),       // global_addr[31:0]
             (unsigned)((ga >> 32) & 0x1FFFFFFull) | (2u << 30) };  // type=2
  v8i g1 = { (int)flags_w0,                        // mask=0|data_size|pad fields
             (int)(0xFFFFu << 16),                 // tensor_dim0 lo16 (0x3FFFFFFF)
             (int)(0x3FFFu | (0xFFFFu << 16)),     // tensor_dim0 hi | dim1 lo
             (int)(0x3FFFu | (tile_d0 << 16)),     // tensor_dim1 hi | tile_dim0
             (int)(tile_d1 & 0xFFFFu),             // tile_dim1 | tile_dim2=0
             (int)(unsigned)(stride_elems & 0xFFFFFFFFull),   // dim0_stride lo
             (int)(unsigned)((stride_elems >> 32) & 0xFFFFull), // stride hi
             0 };
  v4i gz = { 0, 0, 0, 0 };
#if __has_include(<hip/amd_detail/amd_gfx1250_TDM.h>)
  v8i gz8 = { 0, 0, 0, 0, 0, 0, 0, 0 };
  __builtin_amdgcn_tensor_load_to_lds(g0, g1, gz, gz, gz8, 0);   // clang-23 form
#else
  __builtin_amdgcn_tensor_load_to_lds(g0, g1, gz, gz, 0);        // ROCm 7.2 form
#endif
}

// data_size=2B (1<<16) | pad_enable (1<<20) | pad_interval<<22 | pad_amount<<25
// K: pad 4 DW after every 32 DW (code 4)  -> row stride 36 DW = 72 halves
// V: pad 4 DW after every 16 DW (code 3)  -> row stride 20 DW = 40 halves
#define TDM_KFLAGS ((1u << 16) | (1u << 20) | (4u << 22) | (3u << 25))
#define TDM_VFLAGS ((1u << 16) | (1u << 20) | (3u << 22) | (3u << 25))
constexpr int KSTR = 72;   // halves per key row in LDS
constexpr int VSTR = 40;   // halves per h row in LDS

// ---------------------------------------------------------------------------
// Kernel 0: convert + transpose weights fp32 [C,H] -> f16 [H,C] (x3 matrices)
// ---------------------------------------------------------------------------
__global__ void wconv_kernel(const float* __restrict__ Wq,
                             const float* __restrict__ Wk,
                             const float* __restrict__ Wv,
                             _Float16* __restrict__ wt) {
  int idx = blockIdx.x * blockDim.x + threadIdx.x;
  const int per = Cn * Hn;
  if (idx >= 3 * per) return;
  int m   = idx / per;
  int rem = idx - m * per;
  int c   = rem / Hn;
  int h   = rem - c * Hn;
  const float* W = (m == 0) ? Wq : (m == 1) ? Wk : Wv;
  wt[m * per + h * Cn + c] = (_Float16)W[c * Hn + h];
}

// ---------------------------------------------------------------------------
// Kernel 1: QKV projection (unchanged; 12 WMMAs per 32-wide K chunk, A shared)
// ---------------------------------------------------------------------------
__global__ void __launch_bounds__(128)
proj_kernel(const float* __restrict__ x, const _Float16* __restrict__ wt,
            _Float16* __restrict__ qh, _Float16* __restrict__ kh,
            _Float16* __restrict__ vth) {
  const int lane    = threadIdx.x & 31;
  const int wave    = threadIdx.x >> 5;
  const int rowbase = (blockIdx.x * 4 + wave) * 16;
  const int l15     = lane & 15;
  const bool hi     = lane >= 16;

  v8f acc[12] = {};
  const int arow = rowbase + l15;
  const int kb   = hi ? 16 : 0;

  for (int k0 = 0; k0 < Cn; k0 += 32) {
    const float* xp = x + arow * Cn + k0 + kb;
    v8f f0 = *(const v8f*)xp;
    v8f f1 = *(const v8f*)(xp + 8);
    v16h a;
#pragma unroll
    for (int j = 0; j < 8; ++j) { a[j] = (_Float16)f0[j]; a[8 + j] = (_Float16)f1[j]; }

    const int koff = k0 + (hi ? 16 : 0);
#pragma unroll
    for (int m = 0; m < 3; ++m) {
#pragma unroll
      for (int nt = 0; nt < 4; ++nt) {
        const int hcol = nt * 16 + l15;
        const v16h b = *(const v16h*)(wt + m * (Cn * Hn) + hcol * Cn + koff);
        acc[m * 4 + nt] = __builtin_amdgcn_wmma_f32_16x16x32_f16(
            false, a, false, b, (short)0, acc[m * 4 + nt], false, false);
      }
    }
  }

#pragma unroll
  for (int m = 0; m < 3; ++m) {
#pragma unroll
    for (int nt = 0; nt < 4; ++nt) {
      const int col = nt * 16 + l15;
#pragma unroll
      for (int r = 0; r < 8; ++r) {
        const int row = rowbase + r + (hi ? 8 : 0);
        const _Float16 hv = (_Float16)acc[m * 4 + nt][r];
        if (m == 0) {
          qh[row * Hn + col] = hv;
        } else if (m == 1) {
          kh[row * Hn + col] = hv;
        } else {
          const int bi = row >> 11;
          const int t  = row & (Tn - 1);
          vth[(bi * Hn + col) * Tn + t] = hv;
        }
      }
    }
  }
}

// ---------------------------------------------------------------------------
// Kernel 2: causal flash attention, block-cooperative TDM staging.
// 8 waves = 8 consecutive 16-query tiles of one batch. Wave 0 DMAs each
// 32-key K (32x64) and V (64x32, from transposed V) chunk into LDS via
// tensor_load_to_lds (double buffered, TENSORcnt + barrier), all waves
// compute from LDS: 4 WMMAs for S=QK^T, online softmax, P via LDS C->A
// relayout, 4 WMMAs for O += P V.
// ---------------------------------------------------------------------------
__global__ void __launch_bounds__(256)
attn_kernel(const _Float16* __restrict__ qh, const _Float16* __restrict__ kh,
            const _Float16* __restrict__ vth, float* __restrict__ out) {
  __shared__ __align__(16) _Float16 kbuf[2][32 * KSTR];  // 9216 B
  __shared__ __align__(16) _Float16 vbuf[2][64 * VSTR];  // 10240 B
  __shared__ __align__(32) _Float16 pbuf[8][16 * 32];    // 8192 B

  const int lane  = threadIdx.x & 31;
  const int wave  = threadIdx.x >> 5;
  const int l15   = lane & 15;
  const bool hi   = lane >= 16;
  const int qtile = blockIdx.x * 8 + wave;
  const int qrow0 = qtile * 16;          // global row = b*T + t0
  const int b     = qrow0 >> 11;
  const int t0    = qrow0 & (Tn - 1);
  const int t0blk = (blockIdx.x & 15) * 128;   // 16 blocks per batch
  const int kend_w   = t0 + 16;                // this wave's causal bound
  const int kend_blk = t0blk + 128;            // block's max causal bound

  const _Float16* kg = kh  + (size_t)b * Tn * Hn;  // batch base, row-major
  const _Float16* vg = vth + (size_t)b * Hn * Tn;  // batch base, [H][T]

  // ---- Q in A-layout: two 16x32 A matrices (h=0..31, h=32..63) ----
  const int kb = hi ? 16 : 0;
  v16h aq0, aq1;
  {
    const _Float16* qp = qh + (qrow0 + l15) * Hn;
    v8h q00 = *(const v8h*)(qp + kb);
    v8h q01 = *(const v8h*)(qp + kb + 8);
    v8h q10 = *(const v8h*)(qp + 32 + kb);
    v8h q11 = *(const v8h*)(qp + 32 + kb + 8);
#pragma unroll
    for (int j = 0; j < 8; ++j) {
      aq0[j] = q00[j]; aq0[8 + j] = q01[j];
      aq1[j] = q10[j]; aq1[8 + j] = q11[j];
    }
  }

  float mrow[8], lrow[8];
  v8f acc[4] = {};
#pragma unroll
  for (int r = 0; r < 8; ++r) { mrow[r] = -1e30f; lrow[r] = 0.f; }

  const float sscale = 0.03125f;   // C^-0.5 = 1/32 exactly
  _Float16*   pw     = &pbuf[wave][0];

  // ---- preload chunk 0 (K: 1D 2048 halves; V: 2D 32x64, stride T) ----
  if (wave == 0) {
    tdm_load_tile((unsigned)(uintptr_t)&kbuf[0][0], kg,  TDM_KFLAGS, 2048, 0, 64);
    tdm_load_tile((unsigned)(uintptr_t)&vbuf[0][0], vg,  TDM_VFLAGS, 32, 64,
                  (unsigned long long)Tn);
    __builtin_amdgcn_s_wait_tensorcnt(0);
  }
  __syncthreads();

  for (int kc0 = 0, it = 0; kc0 < kend_blk; kc0 += 32, ++it) {
    const int cur = it & 1;

    // prefetch next chunk into the other buffer while computing this one
    if (wave == 0 && (kc0 + 32) < kend_blk) {
      const int nxt = cur ^ 1;
      tdm_load_tile((unsigned)(uintptr_t)&kbuf[nxt][0],
                    kg + (size_t)(kc0 + 32) * Hn, TDM_KFLAGS, 2048, 0, 64);
      tdm_load_tile((unsigned)(uintptr_t)&vbuf[nxt][0],
                    vg + (kc0 + 32), TDM_VFLAGS, 32, 64,
                    (unsigned long long)Tn);
    }

    if (kc0 < kend_w) {
      const _Float16* kL = &kbuf[cur][0];
      const _Float16* vL = &vbuf[cur][0];

      // ---- two 16x16 S tiles (keys kc0..+15, kc0+16..+31) ----
      v8f s[2];
#pragma unroll
      for (int st = 0; st < 2; ++st) {
        const int key0 = kc0 + st * 16;
        if (key0 < kend_w) {
          const _Float16* kp = kL + (st * 16 + l15) * KSTR + (hi ? 16 : 0);
          v8h b0 = *(const v8h*)kp;          // h = hoff+0..7
          v8h b1 = *(const v8h*)(kp + 8);    // h = hoff+8..15
          v8h b2 = *(const v8h*)(kp + 32);   // h = 32+hoff+0..7
          v8h b3 = *(const v8h*)(kp + 40);   // h = 32+hoff+8..15
          v16h bk0, bk1;
#pragma unroll
          for (int j = 0; j < 8; ++j) {
            bk0[j] = b0[j]; bk0[8 + j] = b1[j];
            bk1[j] = b2[j]; bk1[8 + j] = b3[j];
          }
          v8f z = {};
          z = __builtin_amdgcn_wmma_f32_16x16x32_f16(false, aq0, false, bk0,
                                                     (short)0, z, false, false);
          z = __builtin_amdgcn_wmma_f32_16x16x32_f16(false, aq1, false, bk1,
                                                     (short)0, z, false, false);
          const int keyt = key0 + l15;        // key time index (per lane)
#pragma unroll
          for (int r = 0; r < 8; ++r) {
            const int qt = t0 + r + (hi ? 8 : 0);
            z[r] = (keyt <= qt) ? z[r] * sscale : -1e30f;
          }
          s[st] = z;
        } else {
          v8f z;
#pragma unroll
          for (int r = 0; r < 8; ++r) z[r] = -1e30f;
          s[st] = z;
        }
      }

      // ---- online softmax (rows live in 16-lane halves of C-layout) ----
      float sc[8];
#pragma unroll
      for (int r = 0; r < 8; ++r) {
        float cm = fmaxf(s[0][r], s[1][r]);
#pragma unroll
        for (int off = 1; off < 16; off <<= 1) cm = fmaxf(cm, __shfl_xor(cm, off, 32));
        const float mn = fmaxf(mrow[r], cm);
        sc[r] = __expf(mrow[r] - mn);
        const float p0 = __expf(s[0][r] - mn);
        const float p1 = __expf(s[1][r] - mn);
        float rs = p0 + p1;
#pragma unroll
        for (int off = 1; off < 16; off <<= 1) rs += __shfl_xor(rs, off, 32);
        lrow[r] = lrow[r] * sc[r] + rs;
        mrow[r] = mn;
        const int Mr = r + (hi ? 8 : 0);
        pw[Mr * 32 + l15]      = (_Float16)p0;   // stage P row-major 16x32
        pw[Mr * 32 + 16 + l15] = (_Float16)p1;
      }
#pragma unroll
      for (int nt = 0; nt < 4; ++nt)
#pragma unroll
        for (int r = 0; r < 8; ++r) acc[nt][r] *= sc[r];

      asm volatile("s_wait_dscnt 0" ::: "memory");

      // ---- reload P in A-layout (16x32) ----
      v16h pa;
      {
        const _Float16* pp = pw + l15 * 32;
        v8h p0v = *(const v8h*)(pp + kb);
        v8h p1v = *(const v8h*)(pp + kb + 8);
#pragma unroll
        for (int j = 0; j < 8; ++j) { pa[j] = p0v[j]; pa[8 + j] = p1v[j]; }
      }

      // ---- O += P @ V : 4 N-tiles of 16 h-columns, V from LDS ----
#pragma unroll
      for (int nt = 0; nt < 4; ++nt) {
        const int hcol = nt * 16 + l15;
        const _Float16* vp = vL + hcol * VSTR + (hi ? 16 : 0);
        v8h c0 = *(const v8h*)vp;
        v8h c1 = *(const v8h*)(vp + 8);
        v16h bv;
#pragma unroll
        for (int j = 0; j < 8; ++j) { bv[j] = c0[j]; bv[8 + j] = c1[j]; }
        acc[nt] = __builtin_amdgcn_wmma_f32_16x16x32_f16(false, pa, false, bv,
                                                         (short)0, acc[nt], false, false);
      }
    }

    if (wave == 0) __builtin_amdgcn_s_wait_tensorcnt(0);
    __syncthreads();
  }

  // ---- normalize and store fp32 output [B,T,H] ----
#pragma unroll
  for (int nt = 0; nt < 4; ++nt) {
    const int hcol = nt * 16 + l15;
#pragma unroll
    for (int r = 0; r < 8; ++r) {
      const int Mr = r + (hi ? 8 : 0);
      out[(qrow0 + Mr) * Hn + hcol] = acc[nt][r] / lrow[r];
    }
  }
}

// ---------------------------------------------------------------------------
// Host launcher
// ---------------------------------------------------------------------------
extern "C" void kernel_launch(void* const* d_in, const int* in_sizes, int n_in,
                              void* d_out, int out_size, void* d_ws, size_t ws_size,
                              hipStream_t stream) {
  (void)in_sizes; (void)n_in; (void)out_size; (void)ws_size;
  const float* x  = (const float*)d_in[0];
  const float* Wq = (const float*)d_in[1];
  const float* Wk = (const float*)d_in[2];
  const float* Wv = (const float*)d_in[3];
  float* out = (float*)d_out;

  // Workspace layout (f16): wt[3*H*C] | qh[B*T*H] | kh[B*T*H] | vth[B*H*T]
  char* ws = (char*)d_ws;
  const size_t wt_bytes  = (size_t)3 * Cn * Hn * sizeof(_Float16);
  const size_t act_bytes = (size_t)Bn * Tn * Hn * sizeof(_Float16);
  _Float16* wt  = (_Float16*)ws;
  _Float16* qh  = (_Float16*)(ws + wt_bytes);
  _Float16* kh  = (_Float16*)(ws + wt_bytes + act_bytes);
  _Float16* vth = (_Float16*)(ws + wt_bytes + 2 * act_bytes);

  {
    const int total = 3 * Cn * Hn;
    wconv_kernel<<<(total + 255) / 256, 256, 0, stream>>>(Wq, Wk, Wv, wt);
  }
  {
    const int blocks = (Bn * Tn) / 64;  // 512
    proj_kernel<<<blocks, 128, 0, stream>>>(x, wt, qh, kh, vth);
  }
  {
    const int blocks = (Bn * Tn) / (16 * 8);  // 256
    attn_kernel<<<blocks, 256, 0, stream>>>(qh, kh, vth, out);
  }
}
